// PointTransformerLayer_38165079392636
// MI455X (gfx1250) — matrix operand order repaired and checked by reading, not compiled, not run on hardware
//
#include <hip/hip_runtime.h>
#include <hip/hip_bf16.h>

#define NP 768
#define DF 64
#define BTW 72          // bf16 staging tile row stride (elements); 144B rows, 16B aligned chunks
#define NWAVES 4        // waves per block (wave32)
#define JSPLIT 4        // j-range split across blockIdx.y (deterministic partials)
#define NJT 48          // 768 / 16 j-tiles
#define JT_PER (NJT / JSPLIT)

typedef __attribute__((ext_vector_type(16))) __bf16 v16bf;
typedef __attribute__((ext_vector_type(8)))  __bf16 v8bf;
typedef __attribute__((ext_vector_type(8)))  float  v8f;
typedef __attribute__((ext_vector_type(2)))  float  v2f;

// Native hardware f32 -> bf16 convert (RNE); compiler packs pairs into
// v_cvt_pk_bf16_f32 where possible.
__device__ __forceinline__ __bf16 f2bf(float f) { return (__bf16)f; }

// Compiler-level ordering + dscnt drain for cross-lane LDS communication
// within a wave (HW keeps one wave's DS ops in order).
__device__ __forceinline__ void wave_fence() {
  __builtin_amdgcn_wave_barrier();
  asm volatile("s_wait_dscnt 0" ::: "memory");
  __builtin_amdgcn_wave_barrier();
}

__device__ __forceinline__ v16bf cat8(v8bf lo, v8bf hi) {
  return __builtin_shufflevector(lo, hi, 0, 1, 2, 3, 4, 5, 6, 7,
                                 8, 9, 10, 11, 12, 13, 14, 15);
}

// A fragment (16x32 bf16) from bf16 LDS tile [16][BTW] via two 16B DS loads.
// ISA A-layout: lane m holds K = kbase + {8h..8h+7, 16+8h..23+8h}.
__device__ __forceinline__ v16bf load_At(const __bf16* bt, int kbase, int lane) {
  const int half = (lane >> 4) & 1;
  const int m = lane & 15;
  const __bf16* p = bt + m * BTW + kbase + 8 * half;
  v8bf lo = *(const v8bf*)(p);
  v8bf hi = *(const v8bf*)(p + 16);
  return cat8(lo, hi);
}

// B fragment (32x16 bf16) from TRANSPOSED bf16 LDS weights wT[n][k]:
// lane holds column n, K = kbase + {8h..8h+7, 16+8h..23+8h}.
__device__ __forceinline__ v16bf load_Bt(const __bf16* wT, int kbase, int nbase, int lane) {
  const int half = (lane >> 4) & 1;
  const int n = nbase + (lane & 15);
  const __bf16* p = wT + n * DF + kbase + 8 * half;
  v8bf lo = *(const v8bf*)(p);
  v8bf hi = *(const v8bf*)(p + 16);
  return cat8(lo, hi);
}

// C/D 16x16 f32 layout -> bf16 LDS tile (lane = column, VGPR v = row v+8h)
__device__ __forceinline__ void store_cd_bf16(__bf16* bt, const v8f* acc, int lane) {
  const int half = (lane >> 4) & 1;
  const int n0 = lane & 15;
#pragma unroll
  for (int nt = 0; nt < 4; ++nt)
#pragma unroll
    for (int v = 0; v < 8; ++v)
      bt[(v + 8 * half) * BTW + nt * 16 + n0] = f2bf(acc[nt][v]);
}

// One 16x64 x 64x64 layer: 4 N-tiles x 2 K-steps, B frags streamed from LDS.
__device__ __forceinline__ void mma64(v8f* acc, v16bf a0, v16bf a1,
                                      const __bf16* wT, const float* bc,
                                      bool relu, int lane) {
#pragma unroll
  for (int nt = 0; nt < 4; ++nt) {
    v8f c = {0.f, 0.f, 0.f, 0.f, 0.f, 0.f, 0.f, 0.f};
    c = __builtin_amdgcn_wmma_f32_16x16x32_bf16(false, a0, false,
            load_Bt(wT, 0, nt * 16, lane), (short)0, c, false, false);
    c = __builtin_amdgcn_wmma_f32_16x16x32_bf16(false, a1, false,
            load_Bt(wT, 32, nt * 16, lane), (short)0, c, false, false);
#pragma unroll
    for (int v = 0; v < 8; ++v) {
      float x = c[v] + bc[nt];
      c[v] = relu ? fmaxf(x, 0.f) : x;
    }
    acc[nt] = c;
  }
}

// ---------------- setup: per-point projections phi/psi/alpha ----------------
__global__ void proj_kernel(const float* __restrict__ x,
                            const float* __restrict__ phiw, const float* __restrict__ phib,
                            const float* __restrict__ psiw, const float* __restrict__ psib,
                            const float* __restrict__ alw,  const float* __restrict__ alb,
                            float* __restrict__ ws) {
  int idx = blockIdx.x * blockDim.x + threadIdx.x;
  if (idx >= 3 * NP * DF) return;
  const int mat = idx / (NP * DF);
  const int rem = idx % (NP * DF);
  const int i = rem / DF, c = rem % DF;
  const float* w; const float* b;
  if (mat == 0)      { w = phiw; b = phib; }
  else if (mat == 1) { w = psiw; b = psib; }
  else               { w = alw;  b = alb;  }
  float s = b[c];
  const float* xr = x + i * DF;
#pragma unroll 8
  for (int k = 0; k < DF; ++k) s += xr[k] * w[k * DF + c];
  ws[idx] = s;
}

// ---------------- main fused pair-tile kernel ----------------
__global__ __launch_bounds__(NWAVES * 32)
void pt_layer_kernel(const float* __restrict__ p,
                     const float* __restrict__ phi_x,
                     const float* __restrict__ psi_x,
                     const float* __restrict__ alpha_x,
                     const float* __restrict__ pe1_w,
                     const float* __restrict__ pe1_b,
                     const float* __restrict__ pe2_w,
                     const float* __restrict__ pe2_b,
                     const float* __restrict__ g1_w,
                     const float* __restrict__ g1_b,
                     const float* __restrict__ g2_w,
                     const float* __restrict__ g2_b,
                     float* __restrict__ partial) {
  __shared__ __align__(16) __bf16 wTpe2[DF * DF];  // transposed: [n][k]
  __shared__ __align__(16) __bf16 wTg1[DF * DF];
  __shared__ __align__(16) __bf16 wTg2[DF * DF];
  __shared__ __align__(16) __bf16 btile_s[NWAVES][16 * BTW];

  const int tid = threadIdx.x;
  for (int t = tid; t < DF * DF; t += blockDim.x) {
    const int k = t / DF, n = t % DF;
    const int tr = n * DF + k;
    wTpe2[tr] = f2bf(pe2_w[t]);
    wTg1[tr]  = f2bf(g1_w[t]);
    wTg2[tr]  = f2bf(g2_w[t]);
  }
  __syncthreads();

  const int wave = tid >> 5;
  const int lane = tid & 31;
  const int i = blockIdx.x * NWAVES + wave;
  const int js = blockIdx.y;
  __bf16* bt = &btile_s[wave][0];

  const int half = lane >> 4;
  const int n0 = lane & 15;
  const int m = n0;                    // A-matrix row owned by this lane

  // Hoisted per-lane constants (uniform over the j-loop)
  float b1c[4], b2c[4], g1c[4], g2c[4];
  v2f Bp1[4];                          // f32 B frags of pe1_w (4x16 per N-tile)
#pragma unroll
  for (int nt = 0; nt < 4; ++nt) {
    const int c = nt * 16 + n0;
    b1c[nt] = pe1_b[c];
    b2c[nt] = pe2_b[c];
    g1c[nt] = g1_b[c];
    g2c[nt] = g2_b[c];
    Bp1[nt][0] = pe1_w[(half ? 2 : 0) * DF + c];   // K = 2h
    Bp1[nt][1] = half ? 0.f : pe1_w[DF + c];       // K = 2h+1 (K=3 zero pad)
  }
  // phi_x[i] channels pre-gathered in A-fragment order
  float phiA0[16], phiA1[16];
  {
    const float* ph = phi_x + i * DF;
#pragma unroll
    for (int e = 0; e < 16; ++e) {
      const int k = e + 8 * half + 8 * (e >> 3);
      phiA0[e] = ph[k];
      phiA1[e] = ph[32 + k];
    }
  }

  const float pi0 = p[i * 3 + 0], pi1 = p[i * 3 + 1], pi2 = p[i * 3 + 2];
  float yacc[4] = {0.f, 0.f, 0.f, 0.f};

  for (int jt = js * JT_PER; jt < js * JT_PER + JT_PER; ++jt) {
    const int j0 = jt * 16;
    if (jt + 1 < NJT)                       // global_prefetch_b8 next value rows
      __builtin_prefetch(alpha_x + (j0 + 16) * DF, 0, 1);

    // ---- Stage A: t = relu(dp @ pe1_w + pe1_b) via V_WMMA_F32_16X16X4_F32 ----
    v8f t4[4];
    {
      const int j = j0 + m;
      const float d0 = pi0 - p[j * 3 + 0];
      const float d1 = pi1 - p[j * 3 + 1];
      const float d2 = pi2 - p[j * 3 + 2];
      v2f ap;
      ap[0] = half ? d2 : d0;               // K = 2h
      ap[1] = half ? 0.f : d1;              // K = 2h+1 (K=3 zero pad)
#pragma unroll
      for (int nt = 0; nt < 4; ++nt) {
        v8f c = {0.f, 0.f, 0.f, 0.f, 0.f, 0.f, 0.f, 0.f};
        c = __builtin_amdgcn_wmma_f32_16x16x4_f32(false, ap, false, Bp1[nt],
                                                  (short)0, c, false, false);
#pragma unroll
        for (int v = 0; v < 8; ++v) c[v] = fmaxf(c[v] + b1c[nt], 0.f);
        t4[nt] = c;
      }
    }
    store_cd_bf16(bt, t4, lane);            // transpose C/D -> A via LDS
    wave_fence();

    // ---- pe = t @ pe2_w + pe2_b ----
    v16bf a0 = load_At(bt, 0, lane);
    v16bf a1 = load_At(bt, 32, lane);
    v8f pe[4];
    mma64(pe, a0, a1, wTpe2, b2c, false, lane);
    wave_fence();                           // everyone done reading t
    store_cd_bf16(bt, pe, lane);            // bf16 copy of pe for h-build
    wave_fence();

    // ---- h = phi_x[i] - psi_x[j] + pe, built directly in A-fragment order ----
    {
      const float* pr = psi_x + (size_t)(j0 + m) * DF;
      const __bf16* prow = bt + m * BTW + 8 * half;
      v8bf pc0 = *(const v8bf*)(prow);        // pe at K =  8h.. 8h+7
      v8bf pc1 = *(const v8bf*)(prow + 16);   // pe at K = 16+8h..
      v8bf pc2 = *(const v8bf*)(prow + 32);   // pe at K = 32+8h..
      v8bf pc3 = *(const v8bf*)(prow + 48);   // pe at K = 48+8h..
      v16bf h0, h1;
#pragma unroll
      for (int e = 0; e < 16; ++e) {
        const int k = e + 8 * half + 8 * (e >> 3);
        const float pe0 = (float)((e < 8) ? pc0[e & 7] : pc1[e & 7]);
        const float pe1 = (float)((e < 8) ? pc2[e & 7] : pc3[e & 7]);
        h0[e] = f2bf(phiA0[e] - pr[k]      + pe0);
        h1[e] = f2bf(phiA1[e] - pr[32 + k] + pe1);
      }
      a0 = h0; a1 = h1;
    }

    // ---- u = relu(h @ g1_w + g1_b) ----
    v8f u[4];
    mma64(u, a0, a1, wTg1, g1c, true, lane);
    wave_fence();                           // done reading pe copy
    store_cd_bf16(bt, u, lane);
    wave_fence();

    // ---- g = u @ g2_w + g2_b ----
    a0 = load_At(bt, 0, lane);
    a1 = load_At(bt, 32, lane);
    v8f g[4];
    mma64(g, a0, a1, wTg2, g2c, false, lane);

    // ---- per-pair channel softmax + weighted j-sum, all in registers ----
    // Row (v + 8h) channels live across the 16 lanes of this half x 4 N-tiles.
#pragma unroll
    for (int v = 0; v < 8; ++v) {
      float mv = fmaxf(fmaxf(g[0][v], g[1][v]), fmaxf(g[2][v], g[3][v]));
#pragma unroll
      for (int s = 1; s <= 8; s <<= 1) mv = fmaxf(mv, __shfl_xor(mv, s, 32));
      float e0 = __expf(g[0][v] - mv);
      float e1 = __expf(g[1][v] - mv);
      float e2 = __expf(g[2][v] - mv);
      float e3 = __expf(g[3][v] - mv);
      float sum = (e0 + e1) + (e2 + e3);
#pragma unroll
      for (int s = 1; s <= 8; s <<= 1) sum += __shfl_xor(sum, s, 32);
      const float is = 1.f / sum;
      const size_t j = (size_t)(j0 + v + 8 * half);
      const float* ar = alpha_x + j * DF + n0;
      yacc[0] += e0 * is * (ar[0]  + pe[0][v]);
      yacc[1] += e1 * is * (ar[16] + pe[1][v]);
      yacc[2] += e2 * is * (ar[32] + pe[2][v]);
      yacc[3] += e3 * is * (ar[48] + pe[3][v]);
    }
    wave_fence();                           // before next tile reuses btile
  }

  // combine the two half-wave row groups; deterministic partial store
#pragma unroll
  for (int nt = 0; nt < 4; ++nt) yacc[nt] += __shfl_xor(yacc[nt], 16, 32);
  if (half == 0) {
#pragma unroll
    for (int nt = 0; nt < 4; ++nt)
      partial[((size_t)js * NP + i) * DF + nt * 16 + n0] = yacc[nt];
  }
}

// ---------------- fixed-order reduction over j-splits ----------------
__global__ void reduce_kernel(const float* __restrict__ partial, float* __restrict__ out) {
  int idx = blockIdx.x * blockDim.x + threadIdx.x;
  if (idx >= NP * DF) return;
  float s = 0.f;
#pragma unroll
  for (int js = 0; js < JSPLIT; ++js) s += partial[(size_t)js * NP * DF + idx];
  out[idx] = s;
}

extern "C" void kernel_launch(void* const* d_in, const int* in_sizes, int n_in,
                              void* d_out, int out_size, void* d_ws, size_t ws_size,
                              hipStream_t stream) {
  const float* x       = (const float*)d_in[0];
  const float* p       = (const float*)d_in[1];
  const float* phi_w   = (const float*)d_in[2];
  const float* phi_b   = (const float*)d_in[3];
  const float* psi_w   = (const float*)d_in[4];
  const float* psi_b   = (const float*)d_in[5];
  const float* alpha_w = (const float*)d_in[6];
  const float* alpha_b = (const float*)d_in[7];
  const float* g1_w    = (const float*)d_in[8];
  const float* g1_b    = (const float*)d_in[9];
  const float* g2_w    = (const float*)d_in[10];
  const float* g2_b    = (const float*)d_in[11];
  const float* pe1_w   = (const float*)d_in[12];
  const float* pe1_b   = (const float*)d_in[13];
  const float* pe2_w   = (const float*)d_in[14];
  const float* pe2_b   = (const float*)d_in[15];
  (void)in_sizes; (void)n_in; (void)out_size; (void)ws_size;

  float* ws      = (float*)d_ws;
  float* phi_x   = ws;                    // NP*DF
  float* psi_x   = ws + NP * DF;          // NP*DF
  float* alpha_x = ws + 2 * NP * DF;      // NP*DF
  float* partial = ws + 3 * NP * DF;      // JSPLIT * NP * DF
  float* out     = (float*)d_out;

  proj_kernel<<<(3 * NP * DF + 255) / 256, 256, 0, stream>>>(
      x, phi_w, phi_b, psi_w, psi_b, alpha_w, alpha_b, ws);

  dim3 grid(NP / NWAVES, JSPLIT);
  pt_layer_kernel<<<grid, NWAVES * 32, 0, stream>>>(
      p, phi_x, psi_x, alpha_x, pe1_w, pe1_b, pe2_w, pe2_b,
      g1_w, g1_b, g2_w, g2_b, partial);

  reduce_kernel<<<(NP * DF + 255) / 256, 256, 0, stream>>>(partial, out);
}
